// Attention_39256001085948
// MI455X (gfx1250) — compile-verified, matrix operand order checked by reading, not run on hardware
//
#include <hip/hip_runtime.h>
#include <hip/hip_bf16.h>

typedef __bf16 bf16;
typedef __attribute__((ext_vector_type(16))) __bf16 v16bf;
typedef __attribute__((ext_vector_type(8)))  float  v8f;

#define WMMA_BF16(a, b, c) \
  __builtin_amdgcn_wmma_f32_16x16x32_bf16(false, (a), false, (b), (short)0, (c), false, false)

// Generic pointer to an LDS object: low 32 bits are the wave-relative LDS byte
// offset (ISA 10.2: LDS aperture addresses translate as addr[31:0]).
__device__ __forceinline__ unsigned lds_addr(const void* p) {
  return (unsigned)(size_t)p;
}

// CDNA5 async copy: 16B per lane, global -> LDS, tracked by ASYNCcnt.
__device__ __forceinline__ void async_b128(unsigned lds_off, const bf16* g) {
  asm volatile("global_load_async_to_lds_b128 %0, %1, off"
               :: "v"(lds_off), "v"(g) : "memory");
}
#define WAIT_ASYNC(n) asm volatile("s_wait_asynccnt " #n ::: "memory")

// ---------------------------------------------------------------------------
// WMMA fragment loader (ISA 7.12.2, 16-bit A 16x32 / B 32x16 layouts).
// lanes 0-15: K = [k0..k0+7] and [k0+16..k0+23]; lanes 16-31: +8.
// Works for global or LDS base (addrspace inferred; LDS -> ds_load_b128).
// ---------------------------------------------------------------------------
__device__ __forceinline__ v16bf load_frag(const bf16* __restrict__ base,
                                           int ld, int row0, int k0, int lane) {
  const int half8 = (lane >> 4) << 3;
  const bf16* r = base + (size_t)(row0 + (lane & 15)) * ld + k0 + half8;
  union { uint4 u[2]; v16bf v; } f;
  f.u[0] = *(const uint4*)(r);
  f.u[1] = *(const uint4*)(r + 16);
  return f.v;
}

// ---------------------------------------------------------------------------
// Kernel 1: f32 -> bf16 cast
// ---------------------------------------------------------------------------
__global__ void cast_f32_to_bf16(const float* __restrict__ src,
                                 bf16* __restrict__ dst, int n) {
  int i = blockIdx.x * blockDim.x + threadIdx.x;
  if (i < n) dst[i] = (bf16)src[i];
}

// ---------------------------------------------------------------------------
// Shared GEMM core: C[128x128] tile, K-step 32, async-LDS double buffered.
// 256 thr = 8 waves (4 M x 2 N); each wave: 2x4 16x16 accumulators.
// LDS tiles padded to 40-element pitch (80B -> conflict-free 16-row frags).
// ---------------------------------------------------------------------------
__device__ __forceinline__ void gemm_core(
    const bf16* __restrict__ Xg, const bf16* __restrict__ Wg, int Kdim,
    bf16 (&As)[2][128][40], bf16 (&Bs)[2][128][40],
    int mblk, int nblk, v8f (&acc)[2][4]) {
  const int tid  = threadIdx.x;
  const int lane = tid & 31;
  const int wave = tid >> 5;
  const int wm   = (wave & 3) * 32;
  const int wn   = (wave >> 2) * 64;
  const int srow = tid >> 2;        // 0..63 : two rows (srow, srow+64) per thread
  const int scol = (tid & 3) * 8;   // 16B chunk within the 32-element k-slice

  const bf16* Ag = Xg + (size_t)(mblk + srow) * Kdim + scol;
  const bf16* Bg = Wg + (size_t)(nblk + srow) * Kdim + scol;

  // prologue: stage 0
  async_b128(lds_addr(&As[0][srow][scol]),      Ag);
  async_b128(lds_addr(&As[0][srow + 64][scol]), Ag + (size_t)64 * Kdim);
  async_b128(lds_addr(&Bs[0][srow][scol]),      Bg);
  async_b128(lds_addr(&Bs[0][srow + 64][scol]), Bg + (size_t)64 * Kdim);

  const int steps = Kdim >> 5;
  for (int i = 0; i < steps; ++i) {
    if (i + 1 < steps) {                      // issue next stage, then wait for
      const int nb = (i + 1) & 1;             // current (in-order completion)
      const int k0 = (i + 1) << 5;
      async_b128(lds_addr(&As[nb][srow][scol]),      Ag + k0);
      async_b128(lds_addr(&As[nb][srow + 64][scol]), Ag + k0 + (size_t)64 * Kdim);
      async_b128(lds_addr(&Bs[nb][srow][scol]),      Bg + k0);
      async_b128(lds_addr(&Bs[nb][srow + 64][scol]), Bg + k0 + (size_t)64 * Kdim);
      WAIT_ASYNC(0x4);
    } else {
      WAIT_ASYNC(0x0);
    }
    __syncthreads();                          // whole tile resident

    const int b = i & 1;
    v16bf a0 = load_frag(&As[b][0][0], 40, wm,      0, lane);
    v16bf a1 = load_frag(&As[b][0][0], 40, wm + 16, 0, lane);
    v16bf b0 = load_frag(&Bs[b][0][0], 40, wn,      0, lane);
    v16bf b1 = load_frag(&Bs[b][0][0], 40, wn + 16, 0, lane);
    v16bf b2 = load_frag(&Bs[b][0][0], 40, wn + 32, 0, lane);
    v16bf b3 = load_frag(&Bs[b][0][0], 40, wn + 48, 0, lane);
    acc[0][0] = WMMA_BF16(a0, b0, acc[0][0]);
    acc[1][0] = WMMA_BF16(a1, b0, acc[1][0]);
    acc[0][1] = WMMA_BF16(a0, b1, acc[0][1]);
    acc[1][1] = WMMA_BF16(a1, b1, acc[1][1]);
    acc[0][2] = WMMA_BF16(a0, b2, acc[0][2]);
    acc[1][2] = WMMA_BF16(a1, b2, acc[1][2]);
    acc[0][3] = WMMA_BF16(a0, b3, acc[0][3]);
    acc[1][3] = WMMA_BF16(a1, b3, acc[1][3]);
    __syncthreads();                          // reads drained before next overwrite
  }
}

// ---------------------------------------------------------------------------
// Kernel 2: QKV projection (M=4096, N=3072, K=1024); scatters into per-head
// Q[B,H,N,D], K[B,H,N,D], Vt[B,H,D,N] (bf16) so attention frags are contiguous.
// ---------------------------------------------------------------------------
__global__ __launch_bounds__(256) void gemm_qkv(
    const bf16* __restrict__ X, const bf16* __restrict__ W,
    bf16* __restrict__ Q, bf16* __restrict__ K, bf16* __restrict__ Vt) {
  __shared__ __align__(16) bf16 As[2][128][40];
  __shared__ __align__(16) bf16 Bs[2][128][40];
  const int lane = threadIdx.x & 31;
  const int wave = threadIdx.x >> 5;
  const int mblk = blockIdx.x * 128, nblk = blockIdx.y * 128;

  v8f acc[2][4] = {};
  gemm_core(X, W, 1024, As, Bs, mblk, nblk, acc);

  const int m0 = mblk + (wave & 3) * 32;
  const int n0 = nblk + (wave >> 2) * 64;
  const int half8 = (lane >> 4) << 3;
  const int col_l = lane & 15;
  for (int i = 0; i < 2; ++i)
    for (int j = 0; j < 4; ++j)
      for (int v = 0; v < 8; ++v) {
        int m = m0 + i * 16 + v + half8;           // flat token b*2048+tok
        int r = n0 + j * 16 + col_l;               // [0,3072)
        bf16 bv = (bf16)acc[i][j][v];
        int sel = r >> 10, c = r & 1023, h = c >> 6, d = c & 63;
        int bb = m >> 11, tok = m & 2047;
        size_t bh = (size_t)bb * 16 + h;
        if (sel == 0)      Q [(bh * 2048 + tok) * 64 + d]    = bv;
        else if (sel == 1) K [(bh * 2048 + tok) * 64 + d]    = bv;
        else               Vt[(bh * 64   + d)   * 2048 + tok] = bv;
      }
}

// ---------------------------------------------------------------------------
// Kernel 3: flash attention. One (b,h) per blockIdx.x, 128 query rows per
// blockIdx.y; 8 waves x 16 rows. Keys streamed 32 at a time through async-LDS
// (K tile 32x64, V^T tile 64x32 shared by all 8 waves -> 8x load reuse).
// ---------------------------------------------------------------------------
__device__ __forceinline__ float rmax16(float x) {
  for (int off = 1; off < 16; off <<= 1) x = fmaxf(x, __shfl_xor(x, off, 32));
  return x;
}
__device__ __forceinline__ float rsum16(float x) {
  for (int off = 1; off < 16; off <<= 1) x += __shfl_xor(x, off, 32);
  return x;
}

__global__ __launch_bounds__(256) void flash_attn(
    const bf16* __restrict__ Q, const bf16* __restrict__ K,
    const bf16* __restrict__ Vt, bf16* __restrict__ O) {
  __shared__ __align__(16) bf16 Ks[2][32][72];   // keys x d   (pitch-padded)
  __shared__ __align__(16) bf16 Vs[2][64][40];   // d x keys   (pitch-padded)
  __shared__ __align__(16) bf16 pbuf[8][16][40]; // per-wave P staging

  const int tid  = threadIdx.x;
  const int lane = tid & 31;
  const int wave = tid >> 5;
  const int bh   = blockIdx.x;
  const int q0   = blockIdx.y * 128 + wave * 16;
  const float scale = 0.125f;                    // 1/sqrt(64)

  const bf16* Qh = Q  + (size_t)bh * 2048 * 64;
  const bf16* Kh = K  + (size_t)bh * 2048 * 64;
  const bf16* Vh = Vt + (size_t)bh * 64 * 2048;

  const v16bf qf0 = load_frag(Qh, 64, q0, 0,  lane);
  const v16bf qf1 = load_frag(Qh, 64, q0, 32, lane);

  const int half8 = (lane >> 4) << 3;
  const int col_l = lane & 15;

  // staging coordinates: K tile 32 rows x 128B (8 chunks/row), V 64 x 64B (4/row)
  const int krow = tid >> 3, kcol = (tid & 7) * 8;
  const int vrow = tid >> 2, vcol = (tid & 3) * 8;

  float mrow[8], lrow[8], alpha[8];
  v8f o[4] = {};
  for (int v = 0; v < 8; ++v) { mrow[v] = -3.0e30f; lrow[v] = 0.0f; }

  // prologue: stage keys [0,32)
  async_b128(lds_addr(&Ks[0][krow][kcol]), Kh + (size_t)krow * 64 + kcol);
  async_b128(lds_addr(&Vs[0][vrow][vcol]), Vh + (size_t)vrow * 2048 + vcol);

  for (int j = 0; j < 2048; j += 32) {
    if (j + 32 < 2048) {
      const int nb = ((j >> 5) + 1) & 1;
      async_b128(lds_addr(&Ks[nb][krow][kcol]),
                 Kh + (size_t)(j + 32 + krow) * 64 + kcol);
      async_b128(lds_addr(&Vs[nb][vrow][vcol]),
                 Vh + (size_t)vrow * 2048 + j + 32 + vcol);
      WAIT_ASYNC(0x2);
    } else {
      WAIT_ASYNC(0x0);
    }
    __syncthreads();
    const int b = (j >> 5) & 1;

    // S = Q @ K^T : two 16x16 f32 tiles for this 32-key chunk
    v8f s0 = {}, s1 = {};
    s0 = WMMA_BF16(qf0, load_frag(&Ks[b][0][0], 72, 0,  0,  lane), s0);
    s0 = WMMA_BF16(qf1, load_frag(&Ks[b][0][0], 72, 0,  32, lane), s0);
    s1 = WMMA_BF16(qf0, load_frag(&Ks[b][0][0], 72, 16, 0,  lane), s1);
    s1 = WMMA_BF16(qf1, load_frag(&Ks[b][0][0], 72, 16, 32, lane), s1);

    // online softmax; rows live across 16-lane groups in the C layout
    for (int v = 0; v < 8; ++v) {
      float a  = s0[v] * scale;
      float bb = s1[v] * scale;
      float mx = rmax16(fmaxf(a, bb));
      float mn = fmaxf(mrow[v], mx);
      alpha[v] = __expf(mrow[v] - mn);
      mrow[v]  = mn;
      float pa = __expf(a - mn);
      float pb = __expf(bb - mn);
      lrow[v]  = lrow[v] * alpha[v] + rsum16(pa + pb);
      int rr = v + half8;
      pbuf[wave][rr][col_l]      = (bf16)pa;   // C-layout -> row-major P in LDS
      pbuf[wave][rr][16 + col_l] = (bf16)pb;
    }
    for (int t = 0; t < 4; ++t)
      for (int v = 0; v < 8; ++v) o[t][v] *= alpha[v];

    // wave-private staging: drain DS stores before cross-lane frag reads
    asm volatile("s_wait_dscnt 0x0" ::: "memory");

    // O += P (16x32, A-layout from LDS) @ V (32x64)
    v16bf pf = load_frag(&pbuf[wave][0][0], 40, 0, 0, lane);
    o[0] = WMMA_BF16(pf, load_frag(&Vs[b][0][0], 40, 0,  0, lane), o[0]);
    o[1] = WMMA_BF16(pf, load_frag(&Vs[b][0][0], 40, 16, 0, lane), o[1]);
    o[2] = WMMA_BF16(pf, load_frag(&Vs[b][0][0], 40, 32, 0, lane), o[2]);
    o[3] = WMMA_BF16(pf, load_frag(&Vs[b][0][0], 40, 48, 0, lane), o[3]);
    __syncthreads();
  }

  const int b = bh >> 4, h = bh & 15;
  for (int t = 0; t < 4; ++t)
    for (int v = 0; v < 8; ++v) {
      int tok = q0 + v + half8;
      int d   = t * 16 + col_l;
      float val = o[t][v] / lrow[v];
      O[((size_t)b * 2048 + tok) * 1024 + h * 64 + d] = (bf16)val;
    }
}

// ---------------------------------------------------------------------------
// Kernel 4: output projection (M=4096, N=1024, K=1024), f32 out + bias.
// ---------------------------------------------------------------------------
__global__ __launch_bounds__(256) void gemm_out(
    const bf16* __restrict__ A, const bf16* __restrict__ W,
    const float* __restrict__ bias, float* __restrict__ out) {
  __shared__ __align__(16) bf16 As[2][128][40];
  __shared__ __align__(16) bf16 Bs[2][128][40];
  const int lane = threadIdx.x & 31;
  const int wave = threadIdx.x >> 5;
  const int mblk = blockIdx.x * 128, nblk = blockIdx.y * 128;

  v8f acc[2][4] = {};
  gemm_core(A, W, 1024, As, Bs, mblk, nblk, acc);

  const int m0 = mblk + (wave & 3) * 32;
  const int n0 = nblk + (wave >> 2) * 64;
  const int half8 = (lane >> 4) << 3;
  const int col_l = lane & 15;
  for (int i = 0; i < 2; ++i)
    for (int j = 0; j < 4; ++j)
      for (int v = 0; v < 8; ++v) {
        int m = m0 + i * 16 + v + half8;
        int n = n0 + j * 16 + col_l;
        out[(size_t)m * 1024 + n] = acc[i][j][v] + bias[n];
      }
}

// ---------------------------------------------------------------------------
// Host-side launch
// ---------------------------------------------------------------------------
extern "C" void kernel_launch(void* const* d_in, const int* in_sizes, int n_in,
                              void* d_out, int out_size, void* d_ws, size_t ws_size,
                              hipStream_t stream) {
  const float* x      = (const float*)d_in[0];   // [2,2048,1024]
  const float* w_qkv  = (const float*)d_in[1];   // [3072,1024]
  const float* w_proj = (const float*)d_in[2];   // [1024,1024]
  const float* b_proj = (const float*)d_in[3];   // [1024]
  float* out = (float*)d_out;

  char* ws = (char*)d_ws;
  bf16* Xb  = (bf16*)ws; ws += (size_t)4096 * 1024 * 2;
  bf16* Wqb = (bf16*)ws; ws += (size_t)3072 * 1024 * 2;
  bf16* Wpb = (bf16*)ws; ws += (size_t)1024 * 1024 * 2;
  bf16* Qb  = (bf16*)ws; ws += (size_t)32 * 2048 * 64 * 2;
  bf16* Kb  = (bf16*)ws; ws += (size_t)32 * 2048 * 64 * 2;
  bf16* Vtb = (bf16*)ws; ws += (size_t)32 * 64 * 2048 * 2;
  bf16* Ab  = (bf16*)ws; ws += (size_t)4096 * 1024 * 2;

  cast_f32_to_bf16<<<(4096 * 1024 + 255) / 256, 256, 0, stream>>>(x, Xb, 4096 * 1024);
  cast_f32_to_bf16<<<(3072 * 1024 + 255) / 256, 256, 0, stream>>>(w_qkv, Wqb, 3072 * 1024);
  cast_f32_to_bf16<<<(1024 * 1024 + 255) / 256, 256, 0, stream>>>(w_proj, Wpb, 1024 * 1024);

  gemm_qkv  <<<dim3(32, 24), 256, 0, stream>>>(Xb, Wqb, Qb, Kb, Vtb);
  flash_attn<<<dim3(32, 16), 256, 0, stream>>>(Qb, Kb, Vtb, Ab);
  gemm_out  <<<dim3(32,  8), 256, 0, stream>>>(Ab, Wpb, b_proj, out);
}
